// MHSA2D_45372034515579
// MI455X (gfx1250) — matrix operand order repaired and checked by reading, not compiled, not run on hardware
//
#include <hip/hip_runtime.h>
#include <hip/hip_bf16.h>

typedef _Float16 v16h __attribute__((ext_vector_type(16)));
typedef _Float16 v8h  __attribute__((ext_vector_type(8)));
typedef _Float16 v4h  __attribute__((ext_vector_type(4)));
typedef float    v8f  __attribute__((ext_vector_type(8)));
typedef unsigned int u32x4 __attribute__((ext_vector_type(4)));
typedef int          i32x4 __attribute__((ext_vector_type(4)));
typedef int          i32x8 __attribute__((ext_vector_type(8)));

#define HEADS 8
#define DH    32
#define HW    2304   // 48*48
#define CCH   256
#define BT    4

// ---------------- TDM availability (device pass only) ----------------
#if defined(__gfx1250__) && __has_builtin(__builtin_amdgcn_tensor_load_to_lds) && \
    __has_builtin(__builtin_amdgcn_s_wait_tensorcnt)
#define USE_TDM 1
#if defined(__has_include)
#if __has_include(<hip/amd_detail/amd_gfx1250_TDM.h>)
#define TDM_6ARG 1
#endif
#endif
#endif

#ifdef USE_TDM
// 2D f16 tile copy global->LDS via Tensor Data Mover.
//  elems0 = halves per row, rows = #rows, strideElems = global row stride (halves)
//  LDS dest gets padAmount DWORDs inserted after every padInterval chunk
//  (codes: interval 2^(c+1) DWORDs, amount (c+1) DWORDs).
__device__ __forceinline__ void tdm_load_2d_f16(
    void* ldsDst, const _Float16* gsrc,
    unsigned elems0, unsigned rows, unsigned long long strideElems,
    unsigned padIntervalCode, unsigned padAmountCode) {
    unsigned lds_off = (unsigned)(unsigned long long)ldsDst;   // addr[31:0] = LDS offset
    unsigned long long ga = (unsigned long long)gsrc;
    u32x4 g0;
    g0[0] = 1u;                                                // count=1, user mode
    g0[1] = lds_off;                                           // lds_addr
    g0[2] = (unsigned)(ga & 0xffffffffu);                      // global_addr lo
    g0[3] = (unsigned)((ga >> 32) & 0x1ffffffu) | 0x80000000u; // global_addr hi | type=2
    i32x8 g1;
    g1[0] = (int)((1u << 16) |                                 // data_size = 2 bytes
                  (1u << 20) |                                 // pad_enable
                  (padIntervalCode << 22) | (padAmountCode << 25));
    g1[1] = (int)((elems0 & 0xffffu) << 16);                   // tensor_dim0[15:0] @ 63:48
    g1[2] = (int)(((elems0 >> 16) & 0xffffu) |                 // tensor_dim0[31:16]
                  ((rows & 0xffffu) << 16));                   // tensor_dim1[15:0]
    g1[3] = (int)(((rows >> 16) & 0xffffu) |                   // tensor_dim1[31:16]
                  ((elems0 & 0xffffu) << 16));                 // tile_dim0
    g1[4] = (int)(rows & 0xffffu);                             // tile_dim1 (tile_dim2=0)
    g1[5] = (int)(strideElems & 0xffffffffu);                  // tensor_dim0_stride lo
    g1[6] = (int)((strideElems >> 32) & 0xffffu);              // stride hi (dim1_stride=0)
    g1[7] = 0;
    i32x4 z4 = {0, 0, 0, 0};
#ifdef TDM_6ARG
    i32x8 z8 = {0, 0, 0, 0, 0, 0, 0, 0};
    __builtin_amdgcn_tensor_load_to_lds(g0, g1, z4, z4, z8, 0);
#else
    __builtin_amdgcn_tensor_load_to_lds(g0, g1, z4, z4, 0);
#endif
}
#endif  // USE_TDM

// ---- WMMA fragment helpers (CDNA5 16x16x32 f16 layouts, wave32) ----
__device__ __forceinline__ v16h frag_from_chunks(const _Float16* p0, const _Float16* p1) {
    v8h lo = *(const v8h*)p0;
    v8h hi = *(const v8h*)p1;
    return __builtin_shufflevector(lo, hi, 0,1,2,3,4,5,6,7,8,9,10,11,12,13,14,15);
}
__device__ __forceinline__ v16h make_a_frag(const _Float16* row, int b8) {
    return frag_from_chunks(row + b8, row + 16 + b8);
}
__device__ __forceinline__ v16h make_b_frag(const _Float16* row, int b16) {
    return frag_from_chunks(row + b16 / 2, row + 16 + b16 / 2);
}
__device__ __forceinline__ v8f wmma_f16(v16h a, v16h b, v8f c) {
    return __builtin_amdgcn_wmma_f32_16x16x32_f16(false, a, false, b, (short)0, c,
                                                  false, false);
}

// =====================================================================
// Kernel 0: x fp32 [b][c][n]  ->  xt f16 [b][n][c]   (LDS 64x64 transpose)
// grid (4, 36, 4), block 256
// =====================================================================
__global__ __launch_bounds__(256) void xpose_kernel(
    const float* __restrict__ x, _Float16* __restrict__ xt) {
    __shared__ _Float16 tile[64 * 72];
    const int t = threadIdx.x;
    const int bb = blockIdx.x;
    const int n0 = blockIdx.y * 64;
    const int c0 = blockIdx.z * 64;
#pragma unroll
    for (int r = 0; r < 16; ++r) {
        int idx = t + r * 256;
        int c = idx >> 6, n = idx & 63;
        tile[n * 72 + c] = (_Float16)x[((size_t)bb * CCH + c0 + c) * HW + n0 + n];
    }
    __syncthreads();
#pragma unroll
    for (int r = 0; r < 16; ++r) {
        int idx = t + r * 256;
        int n = idx >> 6, c = idx & 63;
        xt[((size_t)bb * HW + n0 + n) * CCH + c0 + c] = tile[n * 72 + c];
    }
}

// =====================================================================
// Shared double-buffered GEMM body:
//   D(128x128) = Wtile(128x256) @ Bsrc(256x128) ; B rows are [n][c] f16
//   A (fp32 weights) staged through registers, B via pipelined TDM.
// =====================================================================
template <typename EPILOGUE>
__device__ __forceinline__ void gemm_core(
    const float* __restrict__ wmat, const _Float16* __restrict__ bsrc_base,
    int o0, EPILOGUE epi,
    _Float16 (*ldsA)[128 * 40], _Float16 (*ldsBt)[128 * 40],
    int t, int wave, int lane15, int b8, int b16) {
    const v8f vz = {0.f,0.f,0.f,0.f,0.f,0.f,0.f,0.f};
    v8f acc[8];
#pragma unroll
    for (int i = 0; i < 8; ++i) acc[i] = vz;

    float4 areg[4];
    auto loadA = [&](int k0) {
#pragma unroll
        for (int r = 0; r < 4; ++r) {
            int idx = t + r * 256;
            int row = idx >> 3, c4 = (idx & 7) * 4;
            areg[r] = *(const float4*)(wmat + (size_t)(o0 + row) * CCH + k0 + c4);
        }
    };
    loadA(0);
#ifdef USE_TDM
    if (wave == 0)
        tdm_load_2d_f16(ldsBt[0], bsrc_base, 32, 128, CCH, /*16DW*/3, /*4DW*/3);
#endif
    for (int ks = 0; ks < 8; ++ks) {
        const int cur = ks & 1;
#pragma unroll
        for (int r = 0; r < 4; ++r) {                    // commit staged A regs
            int idx = t + r * 256;
            int row = idx >> 3, c4 = (idx & 7) * 4;
            v4h hv = {(_Float16)areg[r].x, (_Float16)areg[r].y,
                      (_Float16)areg[r].z, (_Float16)areg[r].w};
            *(v4h*)&ldsA[cur][row * 40 + c4] = hv;
        }
        if (ks + 1 < 8) loadA((ks + 1) * 32);            // overlap next A fetch
#ifdef USE_TDM
        if (wave == 0) {
            if (ks + 1 < 8) {
                tdm_load_2d_f16(ldsBt[cur ^ 1], bsrc_base + (ks + 1) * 32,
                                32, 128, CCH, 3, 3);
                __builtin_amdgcn_s_wait_tensorcnt(1);    // current done, next in flight
            } else {
                __builtin_amdgcn_s_wait_tensorcnt(0);
            }
        }
#else
#pragma unroll
        for (int r = 0; r < 2; ++r) {
            int idx = t + r * 256;
            int row = idx >> 2, c8 = (idx & 3) * 8;
            *(v8h*)&ldsBt[cur][row * 40 + c8] =
                *(const v8h*)(bsrc_base + (size_t)row * CCH + ks * 32 + c8);
        }
#endif
        __syncthreads();
        v16h a = make_a_frag(&ldsA[cur][(wave * 16 + lane15) * 40], b8);
#pragma unroll
        for (int nt = 0; nt < 8; ++nt) {
            v16h bfr = make_b_frag(&ldsBt[cur][(nt * 16 + lane15) * 40], b16);
            acc[nt] = wmma_f16(a, bfr, acc[nt]);
        }
        __syncthreads();
    }
    epi(acc);
}

// =====================================================================
// Kernel 1: qkv = w_qkv(768x256) @ x_b + bias ; q scaled
//   q,k stored [b][part][h][n][d] ; v stored transposed [b][2][h][d][n]
// grid (4, 6, 18), block 256
// =====================================================================
__global__ __launch_bounds__(256) void qkv_gemm_kernel(
    const _Float16* __restrict__ xt, const float* __restrict__ w_qkv,
    const float* __restrict__ b_qkv, _Float16* __restrict__ qkv_out) {
    __shared__ __align__(16) _Float16 ldsA[2][128 * 40];
    __shared__ __align__(16) _Float16 ldsBt[2][128 * 40];

    const int t = threadIdx.x;
    const int wave = t >> 5, lane = t & 31, lane15 = lane & 15;
    const int b8 = (lane >> 4) * 8, b16 = (lane >> 4) * 16;
    const int bb = blockIdx.x;
    const int o0 = blockIdx.y * 128;
    const int n0 = blockIdx.z * 128;

    const _Float16* bsrc = xt + ((size_t)bb * HW + n0) * CCH;

    gemm_core(w_qkv, bsrc, o0,
        [&](v8f (&acc)[8]) {
            const float qscale = 0.17677669529663689f;  // 32^-0.5
#pragma unroll
            for (int r = 0; r < 8; ++r) {
                int o = o0 + wave * 16 + r + b8;        // C layout: M = r + 8*hi
                float bias = b_qkv[o];
                int part = o >> 8, rem = o & 255, h = rem >> 5, d = rem & 31;
                float scl = (part == 0) ? qscale : 1.0f;
                _Float16* base = qkv_out +
                    (((size_t)bb * 3 + part) * HEADS + h) * (size_t)HW * DH;
#pragma unroll
                for (int nt = 0; nt < 8; ++nt) {
                    int n = n0 + nt * 16 + lane15;
                    float v = (acc[nt][r] + bias) * scl;
                    if (part == 2) base[(size_t)d * HW + n] = (_Float16)v;  // V:[d][n]
                    else           base[(size_t)n * DH + d] = (_Float16)v;  // Q,K:[n][d]
                }
            }
        },
        ldsA, ldsBt, t, wave, lane15, b8, b16);
}

// =====================================================================
// Kernel 2: streaming-softmax attention; out -> attn [b][n][c] f16
// grid (32, 18), block 256 (8 waves x 16 query rows); pipelined K/V TDM
// =====================================================================
__global__ __launch_bounds__(256) void attn_kernel(
    const _Float16* __restrict__ qkv, _Float16* __restrict__ attn_out) {
    __shared__ __align__(16) _Float16 ldsQ[128 * 40];         // [n][d]
    __shared__ __align__(16) _Float16 ldsK[2][128 * 40];      // [key][d]
    __shared__ __align__(16) _Float16 ldsVt[2][32 * 136];     // [d][key]
    __shared__ __align__(16) _Float16 ldsP[8 * 16 * 136];     // per-wave P [m][key]

    const int t = threadIdx.x;
    const int wave = t >> 5, lane = t & 31, lane15 = lane & 15;
    const int b8 = (lane >> 4) * 8, b16 = (lane >> 4) * 16;
    const int bh = blockIdx.x;
    const int bb = bh >> 3, h = bh & 7;
    const int q0 = blockIdx.y * 128;
    const int NB = HW / 128;

    const _Float16* qp  = qkv + (((size_t)bb * 3 + 0) * HEADS + h) * (size_t)HW * DH;
    const _Float16* kp  = qkv + (((size_t)bb * 3 + 1) * HEADS + h) * (size_t)HW * DH;
    const _Float16* vtp = qkv + (((size_t)bb * 3 + 2) * HEADS + h) * (size_t)HW * DH;

#ifdef USE_TDM
    if (wave == 0) {                                   // prologue: Q, K0, V0 (3 ops)
        tdm_load_2d_f16(ldsQ, qp + (size_t)q0 * DH, 32, 128, DH, 3, 3);
        tdm_load_2d_f16(ldsK[0], kp, 32, 128, DH, 3, 3);
        tdm_load_2d_f16(ldsVt[0], vtp, 128, 32, HW, /*64DW*/5, /*4DW*/3);
    }
#else
#pragma unroll
    for (int r = 0; r < 2; ++r) {
        int idx = t + r * 256;
        int row = idx >> 2, c8 = (idx & 3) * 8;
        *(v8h*)&ldsQ[row * 40 + c8] =
            *(const v8h*)(qp + (size_t)(q0 + row) * DH + c8);
    }
#endif

    const v8f vz = {0.f,0.f,0.f,0.f,0.f,0.f,0.f,0.f};
    v8f o0acc = vz, o1acc = vz;
    v16h aq = {};
    float mstat[8], lstat[8];
#pragma unroll
    for (int r = 0; r < 8; ++r) { mstat[r] = -1e30f; lstat[r] = 0.f; }

    _Float16* myP = &ldsP[wave * 16 * 136];

    for (int jb = 0; jb < NB; ++jb) {
        const int cur = jb & 1;
#ifdef USE_TDM
        if (wave == 0) {
            if (jb + 1 < NB) {                         // pipeline next K/V block
                tdm_load_2d_f16(ldsK[cur ^ 1], kp + (size_t)(jb + 1) * 128 * DH,
                                32, 128, DH, 3, 3);
                tdm_load_2d_f16(ldsVt[cur ^ 1], vtp + (jb + 1) * 128,
                                128, 32, HW, 5, 3);
                __builtin_amdgcn_s_wait_tensorcnt(2);  // current (+Q) done
            } else {
                __builtin_amdgcn_s_wait_tensorcnt(0);
            }
        }
#else
        const int kbase = jb * 128;
#pragma unroll
        for (int r = 0; r < 2; ++r) {
            int idx = t + r * 256;
            int row = idx >> 2, c8 = (idx & 3) * 8;
            *(v8h*)&ldsK[cur][row * 40 + c8] =
                *(const v8h*)(kp + (size_t)(kbase + row) * DH + c8);
        }
#pragma unroll
        for (int r = 0; r < 2; ++r) {
            int idx = t + r * 256;
            int row = idx >> 4, c8 = (idx & 15) * 8;
            *(v8h*)&ldsVt[cur][row * 136 + c8] =
                *(const v8h*)(vtp + (size_t)row * HW + kbase + c8);
        }
#endif
        __syncthreads();
        if (jb == 0) aq = make_a_frag(&ldsQ[(wave * 16 + lane15) * 40], b8);

        v8f s[8];                                      // S = Q K^T, 16x128 strip
#pragma unroll
        for (int kt = 0; kt < 8; ++kt) {
            v16h bk = make_b_frag(&ldsK[cur][(kt * 16 + lane15) * 40], b16);
            s[kt] = wmma_f16(aq, bk, vz);
        }

#pragma unroll
        for (int r = 0; r < 8; ++r) {                  // online softmax per row
            float mx = s[0][r];
#pragma unroll
            for (int kt = 1; kt < 8; ++kt) mx = fmaxf(mx, s[kt][r]);
#pragma unroll
            for (int m = 1; m < 16; m <<= 1) mx = fmaxf(mx, __shfl_xor(mx, m, 32));
            float mnew = fmaxf(mstat[r], mx);
            float corr = __expf(mstat[r] - mnew);
            mstat[r] = mnew;
            lstat[r] *= corr;
            o0acc[r] *= corr;
            o1acc[r] *= corr;
            float rsum = 0.f;
#pragma unroll
            for (int kt = 0; kt < 8; ++kt) {
                float p = __expf(s[kt][r] - mnew);
                s[kt][r] = p;
                rsum += p;
            }
#pragma unroll
            for (int m = 1; m < 16; m <<= 1) rsum += __shfl_xor(rsum, m, 32);
            lstat[r] += rsum;
#pragma unroll
            for (int kt = 0; kt < 8; ++kt)             // stage P (C->A relayout)
                myP[(r + b8) * 136 + kt * 16 + lane15] = (_Float16)s[kt][r];
        }

#pragma unroll
        for (int kk = 0; kk < 4; ++kk) {               // O += P V
            v16h ap  = make_a_frag(&myP[lane15 * 136 + kk * 32], b8);
            v16h bv0 = make_b_frag(&ldsVt[cur][lane15 * 136 + kk * 32], b16);
            v16h bv1 = make_b_frag(&ldsVt[cur][(16 + lane15) * 136 + kk * 32], b16);
            o0acc = wmma_f16(ap, bv0, o0acc);
            o1acc = wmma_f16(ap, bv1, o1acc);
        }
        __syncthreads();
    }

#pragma unroll
    for (int r = 0; r < 8; ++r) {                      // store attn [b][n][c]
        float inv = 1.0f / fmaxf(lstat[r], 1e-12f);
        int n = q0 + wave * 16 + r + b8;
        int c0 = h * DH + lane15;
        attn_out[((size_t)bb * HW + n) * CCH + c0]      = (_Float16)(o0acc[r] * inv);
        attn_out[((size_t)bb * HW + n) * CCH + c0 + 16] = (_Float16)(o1acc[r] * inv);
    }
}

// =====================================================================
// Kernel 3: out = w_proj(256x256) @ attn + b_proj   (fp32 out [b][c][n])
// grid (4, 2, 18), block 256
// =====================================================================
__global__ __launch_bounds__(256) void proj_gemm_kernel(
    const _Float16* __restrict__ attn, const float* __restrict__ w_proj,
    const float* __restrict__ b_proj, float* __restrict__ out) {
    __shared__ __align__(16) _Float16 ldsA[2][128 * 40];
    __shared__ __align__(16) _Float16 ldsBt[2][128 * 40];

    const int t = threadIdx.x;
    const int wave = t >> 5, lane = t & 31, lane15 = lane & 15;
    const int b8 = (lane >> 4) * 8, b16 = (lane >> 4) * 16;
    const int bb = blockIdx.x;
    const int o0 = blockIdx.y * 128;
    const int n0 = blockIdx.z * 128;

    const _Float16* bsrc = attn + ((size_t)bb * HW + n0) * CCH;

    gemm_core(w_proj, bsrc, o0,
        [&](v8f (&acc)[8]) {
#pragma unroll
            for (int r = 0; r < 8; ++r) {
                int o = o0 + wave * 16 + r + b8;
                float bias = b_proj[o];
#pragma unroll
                for (int nt = 0; nt < 8; ++nt) {
                    int n = n0 + nt * 16 + lane15;
                    out[((size_t)bb * CCH + o) * HW + n] = acc[nt][r] + bias;
                }
            }
        },
        ldsA, ldsBt, t, wave, lane15, b8, b16);
}

extern "C" void kernel_launch(void* const* d_in, const int* in_sizes, int n_in,
                              void* d_out, int out_size, void* d_ws, size_t ws_size,
                              hipStream_t stream) {
    (void)in_sizes; (void)n_in; (void)out_size; (void)ws_size;
    const float* x      = (const float*)d_in[0];
    const float* w_qkv  = (const float*)d_in[1];
    const float* b_qkv  = (const float*)d_in[2];
    const float* w_proj = (const float*)d_in[3];
    const float* b_proj = (const float*)d_in[4];

    _Float16* ws_xt   = (_Float16*)d_ws;                             // [b][n][c] f16
    _Float16* ws_qkv  = ws_xt + (size_t)BT * HW * CCH;               // q,k:[n][d] v:[d][n]
    _Float16* ws_attn = ws_qkv + (size_t)BT * 3 * HEADS * HW * DH;   // [b][n][c] f16

    xpose_kernel<<<dim3(BT, HW / 64, CCH / 64), 256, 0, stream>>>(x, ws_xt);
    qkv_gemm_kernel<<<dim3(BT, 6, 18), 256, 0, stream>>>(ws_xt, w_qkv, b_qkv, ws_qkv);
    attn_kernel<<<dim3(BT * HEADS, HW / 128), 256, 0, stream>>>(ws_qkv, ws_attn);
    proj_gemm_kernel<<<dim3(BT, 2, 18), 256, 0, stream>>>(ws_attn, w_proj, b_proj,
                                                          (float*)d_out);
}
